// TransformerEncoderBlock_26860725469640
// MI455X (gfx1250) — compile-verified, matrix operand order checked by reading, not run on hardware
//
#include <hip/hip_runtime.h>
#include <hip/hip_fp16.h>

typedef _Float16     v16h  __attribute__((ext_vector_type(16)));
typedef float        v8f   __attribute__((ext_vector_type(8)));
typedef unsigned int u32x4 __attribute__((ext_vector_type(4)));
typedef float        f32x4 __attribute__((ext_vector_type(4)));

union Frag16 { u32x4 u[2]; v16h h; };   // 16 f16 = one WMMA A/B fragment per lane
union H8     { u32x4 u;   _Float16 h[8]; };

#define NEG_BIG (-3.4028235e38f)

// Async DMA global->LDS (CDNA5 GLOBAL_LOAD_ASYNC_TO_LDS_B128, ASYNCcnt-tracked).
__device__ __forceinline__ void async_copy_b128(const _Float16* gptr, void* ldsPtr) {
  unsigned ldsOff = (unsigned)(size_t)ldsPtr;   // ISA: flat->LDS uses addr[31:0]
  asm volatile("global_load_async_to_lds_b128 %0, %1, off"
               :: "v"(ldsOff), "v"(gptr) : "memory");
}
__device__ __forceinline__ void async_wait0() {
  asm volatile("s_wait_asynccnt 0" ::: "memory");
}

// ---------------------------------------------------------------- fp32 -> f16
__global__ void __launch_bounds__(256) cvt_f32_f16(const float* __restrict__ in,
                                                   _Float16* __restrict__ out, int n) {
  int i = blockIdx.x * blockDim.x + threadIdx.x;
  int stride = gridDim.x * blockDim.x;
  for (; i < n; i += stride) out[i] = (_Float16)in[i];
}

// ------------------------------------------- fp32 W[K,N] -> f16 Wt[N,K]
// One-time tiled transpose+convert so GEMM B tiles become straight async DMAs.
__global__ void __launch_bounds__(256) cvt_transpose_w(
    const float* __restrict__ W, _Float16* __restrict__ Wt, int K, int N)
{
  __shared__ _Float16 Ts[64 * 72];
  const int tid = threadIdx.x;
  const int n0 = blockIdx.x * 64, k0 = blockIdx.y * 64;
  #pragma unroll
  for (int i = 0; i < 2; ++i) {
    int idx = tid + i * 256;
    int r = idx >> 3, c = (idx & 7) << 3;          // r = k row, c = n col
    const float* p = W + (size_t)(k0 + r) * N + n0 + c;
    f32x4 lo = *reinterpret_cast<const f32x4*>(p);
    f32x4 hi = *reinterpret_cast<const f32x4*>(p + 4);
    H8 h;
    h.h[0]=(_Float16)lo.x; h.h[1]=(_Float16)lo.y; h.h[2]=(_Float16)lo.z; h.h[3]=(_Float16)lo.w;
    h.h[4]=(_Float16)hi.x; h.h[5]=(_Float16)hi.y; h.h[6]=(_Float16)hi.z; h.h[7]=(_Float16)hi.w;
    *reinterpret_cast<u32x4*>(&Ts[r * 72 + c]) = h.u;
  }
  __syncthreads();
  #pragma unroll
  for (int i = 0; i < 2; ++i) {
    int idx = tid + i * 256;
    int n = idx >> 3, ck = (idx & 7) << 3;         // n = out row, ck = k col
    H8 h;
    #pragma unroll
    for (int j = 0; j < 8; ++j) h.h[j] = Ts[(ck + j) * 72 + n];
    *reinterpret_cast<u32x4*>(Wt + (size_t)(n0 + n) * K + k0 + ck) = h.u;
  }
}

// -------------------------- V half of KV [B,T,2D] -> Vt [B,H,DH,T] (f16)
__global__ void __launch_bounds__(256) transpose_v(
    const _Float16* __restrict__ KV, _Float16* __restrict__ Vt)
{
  constexpr int T = 1024, KVLD = 2048, DH = 64;
  __shared__ _Float16 Ts[64 * 72];
  const int tid = threadIdx.x;
  const int h = blockIdx.y & 15, b = blockIdx.y >> 4;
  const int t0 = blockIdx.x * 64;
  const size_t ibase = ((size_t)b * T + t0) * KVLD + 1024 + h * DH;
  const size_t obase = ((size_t)(b * 16 + h)) * DH * T;
  #pragma unroll
  for (int i = 0; i < 2; ++i) {
    int idx = tid + i * 256;
    int r = idx >> 3, c = (idx & 7) << 3;          // r = t, c = d
    *reinterpret_cast<u32x4*>(&Ts[r * 72 + c]) =
        *reinterpret_cast<const u32x4*>(KV + ibase + (size_t)r * KVLD + c);
  }
  __syncthreads();
  #pragma unroll
  for (int i = 0; i < 2; ++i) {
    int idx = tid + i * 256;
    int d = idx >> 3, ct = (idx & 7) << 3;
    H8 h;
    #pragma unroll
    for (int j = 0; j < 8; ++j) h.h[j] = Ts[(ct + j) * 72 + d];
    *reinterpret_cast<u32x4*>(Vt + obase + (size_t)d * T + t0 + ct) = h.u;
  }
}

// ------------------------------------------------------------------- GEMM
// C[M,N] = A[M,K](f16) @ Wt[N,K](f16, pre-transposed) + bias (+resid, relu).
// 128x128 tile, BK=32, 8 waves x (32x64). Double-buffered LDS: tile k+1 is
// DMA'd by the async engine while tile k's WMMAs run.
__global__ void __launch_bounds__(256) gemm_f16_wmma(
    const _Float16* __restrict__ A, const _Float16* __restrict__ Wt,
    const float* __restrict__ bias, const float* __restrict__ resid,
    _Float16* __restrict__ outH, float* __restrict__ outF,
    int M, int N, int K, int relu)
{
  __shared__ _Float16 As[2][128 * 40];   // [m][k], pad 32->40 (80B rows)
  __shared__ _Float16 Bs[2][128 * 40];   // [n][k]
  const int tid  = threadIdx.x;
  const int m0   = blockIdx.y * 128;
  const int n0   = blockIdx.x * 128;
  const int wid  = tid >> 5, lane = tid & 31;
  const int wm   = wid & 3,  wn   = wid >> 2;
  const int lrow = lane & 15, kg  = lane >> 4;

  auto issue_tile = [&](int kt, int buf) {
    const int k0 = kt << 5;
    #pragma unroll
    for (int i = 0; i < 2; ++i) {
      int idx = tid + i * 256;                    // 512 chunks of 8 per tile
      int r = idx >> 2, c = (idx & 3) << 3;       // r = row, c = k col
      async_copy_b128(A  + (size_t)(m0 + r) * K + k0 + c, &As[buf][r * 40 + c]);
      async_copy_b128(Wt + (size_t)(n0 + r) * K + k0 + c, &Bs[buf][r * 40 + c]);
    }
  };

  v8f acc[2][4];
  #pragma unroll
  for (int mt = 0; mt < 2; ++mt)
    #pragma unroll
    for (int nt = 0; nt < 4; ++nt) acc[mt][nt] = {};

  const int nK = K >> 5;
  issue_tile(0, 0);
  for (int kt = 0; kt < nK; ++kt) {
    const int buf = kt & 1;
    async_wait0();          // tile kt resident (DMA overlapped prior compute)
    __syncthreads();
    if (kt + 1 < nK) issue_tile(kt + 1, buf ^ 1);
    if (kt + 2 < nK) {      // gfx1250 global_prefetch_b8 two tiles ahead
      __builtin_prefetch(A  + (size_t)(m0 + (tid >> 2)) * K + (kt << 5) + 64 + ((tid & 3) << 3), 0, 1);
      __builtin_prefetch(Wt + (size_t)(n0 + (tid >> 2)) * K + (kt << 5) + 64 + ((tid & 3) << 3), 0, 1);
    }

    Frag16 a[2], b[4];
    #pragma unroll
    for (int mt = 0; mt < 2; ++mt) {
      const _Float16* p = &As[buf][(wm * 32 + mt * 16 + lrow) * 40 + kg * 8];
      a[mt].u[0] = *reinterpret_cast<const u32x4*>(p);
      a[mt].u[1] = *reinterpret_cast<const u32x4*>(p + 16);
    }
    #pragma unroll
    for (int nt = 0; nt < 4; ++nt) {
      const _Float16* p = &Bs[buf][(wn * 64 + nt * 16 + lrow) * 40 + kg * 8];
      b[nt].u[0] = *reinterpret_cast<const u32x4*>(p);
      b[nt].u[1] = *reinterpret_cast<const u32x4*>(p + 16);
    }
    #pragma unroll
    for (int mt = 0; mt < 2; ++mt)
      #pragma unroll
      for (int nt = 0; nt < 4; ++nt)
        acc[mt][nt] = __builtin_amdgcn_wmma_f32_16x16x32_f16(
            false, a[mt].h, false, b[nt].h, (short)0, acc[mt][nt], false, false);
  }

  #pragma unroll
  for (int mt = 0; mt < 2; ++mt)
    #pragma unroll
    for (int nt = 0; nt < 4; ++nt) {
      const int gn = n0 + wn * 64 + nt * 16 + lrow;
      const float bv = bias ? bias[gn] : 0.f;
      #pragma unroll
      for (int r = 0; r < 8; ++r) {
        const int gm = m0 + wm * 32 + mt * 16 + r + kg * 8;
        float v = acc[mt][nt][r] + bv;
        if (relu) v = v > 0.f ? v : 0.f;
        const size_t o = (size_t)gm * N + gn;
        if (resid) v += resid[o];
        if (outF) outF[o] = v;
        if (outH) outH[o] = (_Float16)v;
      }
    }
}

// --------------------------------------------------------------- attention
// Block = (b, h, 64 q rows); 4 waves x 16 q rows; streams 64-key blocks with
// online softmax. All tile staging is async-DMA, K/V double-buffered.
// Q f16 [B,T,D]; KV f16 [B,T,2D] (K | V); VtG f16 [B,H,DH,T].
__global__ void __launch_bounds__(128) attn_wmma(
    const _Float16* __restrict__ Q, const _Float16* __restrict__ KV,
    const _Float16* __restrict__ VtG, const int* __restrict__ mask,
    _Float16* __restrict__ ctx)
{
  constexpr int T = 1024, D = 1024, DH = 64, KVLD = 2048;
  __shared__ _Float16 QP[64 * 72];       // Q tile; per-wave region reused for P
  __shared__ _Float16 Ks2[2][64 * 72];   // [key][d]  (B-frag layout for QK^T)
  __shared__ _Float16 Vs2[2][64 * 72];   // [d][key]  (B-frag layout for PV)
  const int tid  = threadIdx.x;
  const int w    = tid >> 5, lane = tid & 31;
  const int lrow = lane & 15, kg = lane >> 4;
  const int h    = blockIdx.y & 15, b = blockIdx.y >> 4;
  const int q0   = blockIdx.x * 64;
  const size_t qbase  = ((size_t)b * T + q0) * D + h * DH;
  const size_t kvbase = ((size_t)b * T) * KVLD + h * DH;
  const size_t vbase  = ((size_t)(b * 16 + h)) * DH * T;

  auto issue_kv = [&](int kb, int buf) {
    const int k0 = kb * 64;
    #pragma unroll
    for (int i = 0; i < 4; ++i) {
      int idx = tid + i * 128;
      int r = idx >> 3, c = (idx & 7) << 3;
      async_copy_b128(KV  + kvbase + (size_t)(k0 + r) * KVLD + c, &Ks2[buf][r * 72 + c]);
      async_copy_b128(VtG + vbase  + (size_t)r * T + k0 + c,      &Vs2[buf][r * 72 + c]);
    }
  };

  #pragma unroll
  for (int i = 0; i < 4; ++i) {          // Q tile: async DMA to LDS
    int idx = tid + i * 128;
    int r = idx >> 3, c = (idx & 7) << 3;
    async_copy_b128(Q + qbase + (size_t)r * D + c, &QP[r * 72 + c]);
  }
  async_wait0();
  __syncthreads();
  Frag16 qf[2];
  #pragma unroll
  for (int c = 0; c < 2; ++c) {          // wave w reads rows 16w..16w+15 only
    const _Float16* p = &QP[(w * 16 + lrow) * 72 + c * 32 + kg * 8];
    qf[c].u[0] = *reinterpret_cast<const u32x4*>(p);
    qf[c].u[1] = *reinterpret_cast<const u32x4*>(p + 16);
  }

  float mrow[8], lsum[8];
  v8f o[4];
  #pragma unroll
  for (int r = 0; r < 8; ++r) { mrow[r] = NEG_BIG; lsum[r] = 0.f; }
  #pragma unroll
  for (int dt = 0; dt < 4; ++dt) o[dt] = {};

  issue_kv(0, 0);
  for (int kb = 0; kb < 16; ++kb) {
    const int buf = kb & 1;
    const int k0 = kb * 64;
    async_wait0();
    __syncthreads();
    if (kb + 1 < 16) issue_kv(kb + 1, buf ^ 1);

    v8f s[4];
    #pragma unroll
    for (int nt = 0; nt < 4; ++nt) {
      s[nt] = {};
      #pragma unroll
      for (int c = 0; c < 2; ++c) {
        Frag16 bf;
        const _Float16* p = &Ks2[buf][(nt * 16 + lrow) * 72 + c * 32 + kg * 8];
        bf.u[0] = *reinterpret_cast<const u32x4*>(p);
        bf.u[1] = *reinterpret_cast<const u32x4*>(p + 16);
        s[nt] = __builtin_amdgcn_wmma_f32_16x16x32_f16(
            false, qf[c].h, false, bf.h, (short)0, s[nt], false, false);
      }
      const int mk = mask[b * T + k0 + nt * 16 + lrow];
      #pragma unroll
      for (int r = 0; r < 8; ++r)
        s[nt][r] = mk ? s[nt][r] * 0.125f : NEG_BIG;   // q/sqrt(64) scaling
    }
    float cand[8], sf[8], rs[8];
    #pragma unroll
    for (int r = 0; r < 8; ++r) {
      float c0 = fmaxf(fmaxf(s[0][r], s[1][r]), fmaxf(s[2][r], s[3][r]));
      c0 = fmaxf(c0, __shfl_xor(c0, 1, 32));
      c0 = fmaxf(c0, __shfl_xor(c0, 2, 32));
      c0 = fmaxf(c0, __shfl_xor(c0, 4, 32));
      c0 = fmaxf(c0, __shfl_xor(c0, 8, 32));
      cand[r] = c0;
    }
    #pragma unroll
    for (int r = 0; r < 8; ++r) {
      float mn = fmaxf(mrow[r], cand[r]);
      sf[r] = __expf(mrow[r] - mn);
      mrow[r] = mn;
      rs[r] = 0.f;
    }
    #pragma unroll
    for (int nt = 0; nt < 4; ++nt)
      #pragma unroll
      for (int r = 0; r < 8; ++r) {
        float p = __expf(s[nt][r] - mrow[r]);
        s[nt][r] = p; rs[r] += p;
      }
    #pragma unroll
    for (int r = 0; r < 8; ++r) {
      float t = rs[r];
      t += __shfl_xor(t, 1, 32);
      t += __shfl_xor(t, 2, 32);
      t += __shfl_xor(t, 4, 32);
      t += __shfl_xor(t, 8, 32);
      lsum[r] = lsum[r] * sf[r] + t;
    }
    #pragma unroll
    for (int dt = 0; dt < 4; ++dt)
      #pragma unroll
      for (int r = 0; r < 8; ++r) o[dt][r] *= sf[r];

    // C-frag -> A-frag re-layout of P through this wave's own LDS region
    // (wave-local write->read: ordered by DScnt, no block barrier needed)
    #pragma unroll
    for (int nt = 0; nt < 4; ++nt)
      #pragma unroll
      for (int r = 0; r < 8; ++r)
        QP[w * 1152 + (r + kg * 8) * 72 + nt * 16 + lrow] = (_Float16)s[nt][r];
    __builtin_amdgcn_wave_barrier();
    Frag16 pf[2];
    #pragma unroll
    for (int c = 0; c < 2; ++c) {
      const _Float16* p = &QP[w * 1152 + lrow * 72 + c * 32 + kg * 8];
      pf[c].u[0] = *reinterpret_cast<const u32x4*>(p);
      pf[c].u[1] = *reinterpret_cast<const u32x4*>(p + 16);
    }
    #pragma unroll
    for (int dt = 0; dt < 4; ++dt)
      #pragma unroll
      for (int c = 0; c < 2; ++c) {
        Frag16 bf;
        const _Float16* p = &Vs2[buf][(dt * 16 + lrow) * 72 + c * 32 + kg * 8];
        bf.u[0] = *reinterpret_cast<const u32x4*>(p);
        bf.u[1] = *reinterpret_cast<const u32x4*>(p + 16);
        o[dt] = __builtin_amdgcn_wmma_f32_16x16x32_f16(
            false, pf[c].h, false, bf.h, (short)0, o[dt], false, false);
      }
  }

  #pragma unroll
  for (int dt = 0; dt < 4; ++dt)
    #pragma unroll
    for (int r = 0; r < 8; ++r) {
      float v = o[dt][r] / lsum[r];
      ctx[((size_t)b * T + q0 + w * 16 + r + kg * 8) * D + h * DH + dt * 16 + lrow] = (_Float16)v;
    }
}

// ------------------------------------------------------------- LayerNorm
// allennlp: g*(x-mean)/(sqrt(unbiased var)+eps)+b ; one block per row (N=1024)
__global__ void __launch_bounds__(256) layernorm_k(
    const float* __restrict__ x, const float* __restrict__ g, const float* __restrict__ bta,
    float* __restrict__ outF, _Float16* __restrict__ outH, int N)
{
  __shared__ float red1[8], red2[8];
  const int row = blockIdx.x, tid = threadIdx.x;
  const f32x4 v = *reinterpret_cast<const f32x4*>(x + (size_t)row * N + tid * 4);
  float s1 = v.x + v.y + v.z + v.w;
  float s2 = v.x * v.x + v.y * v.y + v.z * v.z + v.w * v.w;
  #pragma unroll
  for (int off = 16; off > 0; off >>= 1) {
    s1 += __shfl_xor(s1, off, 32);
    s2 += __shfl_xor(s2, off, 32);
  }
  const int wid = tid >> 5, lane = tid & 31;
  if (lane == 0) { red1[wid] = s1; red2[wid] = s2; }
  __syncthreads();
  if (tid == 0) {
    float a = 0.f, bq = 0.f;
    for (int i = 0; i < 8; ++i) { a += red1[i]; bq += red2[i]; }
    red1[0] = a; red2[0] = bq;
  }
  __syncthreads();
  const float mean = red1[0] / N;
  float var = red2[0] / N - mean * mean;
  var = var * ((float)N / (float)(N - 1));
  const float inv = 1.f / (sqrtf(fmaxf(var, 0.f)) + 1e-6f);
  const float xv[4] = { v.x, v.y, v.z, v.w };
  #pragma unroll
  for (int j = 0; j < 4; ++j) {
    const int c = tid * 4 + j;
    const float ov = g[c] * (xv[j] - mean) * inv + bta[c];
    const size_t o = (size_t)row * N + c;
    if (outF) outF[o] = ov;
    if (outH) outH[o] = (_Float16)ov;
  }
}

// ---------------------------------------------------------------- driver
extern "C" void kernel_launch(void* const* d_in, const int* in_sizes, int n_in,
                              void* d_out, int out_size, void* d_ws, size_t ws_size,
                              hipStream_t stream) {
  (void)in_sizes; (void)n_in; (void)out_size; (void)ws_size;
  constexpr int B = 4, T = 1024, D = 1024, FF = 4096;
  const int M = B * T;

  const float* src  = (const float*)d_in[0];
  const int*   smsk = (const int*)  d_in[1];
  const float* q_w  = (const float*)d_in[2];
  const float* q_b  = (const float*)d_in[3];
  const float* kv_w = (const float*)d_in[4];
  const float* kv_b = (const float*)d_in[5];
  const float* o_w  = (const float*)d_in[6];
  const float* o_b  = (const float*)d_in[7];
  const float* w1   = (const float*)d_in[8];
  const float* b1   = (const float*)d_in[9];
  const float* w2   = (const float*)d_in[10];
  const float* b2   = (const float*)d_in[11];
  const float* n1g  = (const float*)d_in[12];
  const float* n1b  = (const float*)d_in[13];
  const float* n2g  = (const float*)d_in[14];
  const float* n2b  = (const float*)d_in[15];

  char* ws = (char*)d_ws;
  size_t cur = 0;
  auto alloc = [&](size_t bytes) -> char* {
    char* p = ws + cur;
    cur = (cur + bytes + 255) & ~(size_t)255;
    return p;
  };
  _Float16* srcH  = (_Float16*)alloc((size_t)M * D * 2);
  _Float16* qwT   = (_Float16*)alloc((size_t)D * D * 2);        // [N,K]
  _Float16* kvwT  = (_Float16*)alloc((size_t)D * 2 * D * 2);    // [2D,D]
  _Float16* owT   = (_Float16*)alloc((size_t)D * D * 2);
  _Float16* w1T   = (_Float16*)alloc((size_t)D * FF * 2);       // [FF,D]
  _Float16* w2T   = (_Float16*)alloc((size_t)FF * D * 2);       // [D,FF]
  _Float16* qH    = (_Float16*)alloc((size_t)M * D * 2);
  _Float16* kvH   = (_Float16*)alloc((size_t)M * 2 * D * 2);
  _Float16* vtH   = (_Float16*)alloc((size_t)M * D * 2);        // [B,H,DH,T]
  _Float16* ctxH  = (_Float16*)alloc((size_t)M * D * 2);
  _Float16* out1H = (_Float16*)alloc((size_t)M * D * 2);
  _Float16* h1H   = (_Float16*)alloc((size_t)M * FF * 2);
  float*    res1F = (float*)   alloc((size_t)M * D * 4);
  float*    out1F = (float*)   alloc((size_t)M * D * 4);
  float*    res2F = (float*)   alloc((size_t)M * D * 4);

  // activation convert; weights convert+transpose (one-time, reused 32x)
  cvt_f32_f16<<<2048, 256, 0, stream>>>(src, srcH, M * D);
  cvt_transpose_w<<<dim3(D / 64,      D / 64),  256, 0, stream>>>(q_w,  qwT,  D,  D);
  cvt_transpose_w<<<dim3(2 * D / 64,  D / 64),  256, 0, stream>>>(kv_w, kvwT, D,  2 * D);
  cvt_transpose_w<<<dim3(D / 64,      D / 64),  256, 0, stream>>>(o_w,  owT,  D,  D);
  cvt_transpose_w<<<dim3(FF / 64,     D / 64),  256, 0, stream>>>(w1,   w1T,  D,  FF);
  cvt_transpose_w<<<dim3(D / 64,      FF / 64), 256, 0, stream>>>(w2,   w2T,  FF, D);

  // Q and KV projections
  gemm_f16_wmma<<<dim3(D / 128, M / 128), 256, 0, stream>>>(
      srcH, qwT, q_b, nullptr, qH, nullptr, M, D, D, 0);
  gemm_f16_wmma<<<dim3(2 * D / 128, M / 128), 256, 0, stream>>>(
      srcH, kvwT, kv_b, nullptr, kvH, nullptr, M, 2 * D, D, 0);

  // V -> [B,H,DH,T] so attention V tiles are straight async DMAs
  transpose_v<<<dim3(T / 64, B * 16), 256, 0, stream>>>(kvH, vtH);

  // flash attention
  attn_wmma<<<dim3(T / 64, B * 16), 128, 0, stream>>>(qH, kvH, vtH, smsk, ctxH);

  // output projection + residual(src) -> LN1
  gemm_f16_wmma<<<dim3(D / 128, M / 128), 256, 0, stream>>>(
      ctxH, owT, o_b, src, nullptr, res1F, M, D, D, 0);
  layernorm_k<<<M, 256, 0, stream>>>(res1F, n1g, n1b, out1F, out1H, D);

  // FFN
  gemm_f16_wmma<<<dim3(FF / 128, M / 128), 256, 0, stream>>>(
      out1H, w1T, b1, nullptr, h1H, nullptr, M, FF, D, 1);
  gemm_f16_wmma<<<dim3(D / 128, M / 128), 256, 0, stream>>>(
      h1H, w2T, b2, out1F, nullptr, res2F, M, D, FF, 0);
  layernorm_k<<<M, 256, 0, stream>>>(res2F, n2g, n2b, (float*)d_out, nullptr, D);
}
